// SampledSoftmaxLayer_24429773980285
// MI455X (gfx1250) — matrix operand order repaired and checked by reading, not compiled, not run on hardware
//
#include <hip/hip_runtime.h>
#include <hip/hip_bf16.h>

typedef __attribute__((ext_vector_type(16))) __bf16 v16bf;
typedef __attribute__((ext_vector_type(8)))  float  v8f;
typedef __attribute__((ext_vector_type(4)))  unsigned int v4u;
typedef __attribute__((ext_vector_type(8)))  int v8i;
typedef __attribute__((ext_vector_type(4)))  int v4i;

#define NUM_CLASSES 100000
#define NUM_SAMPLED 1000
#define DIM 512
#define BATCH 8192
#define NPAD 1008            // 63 tiles of 16
#define NTILES 63
#define LDS_ROW 520          // 512 bf16 + 8 bf16 (16B) TDM pad -> bank-conflict-free

// workspace layout (bytes, all 256-aligned)
#define WSB_OFF 0u            // bf16 [NPAD][DIM]      = 1,032,192 B
#define SB_OFF  1032192u      // f32  [NPAD]           =     4,032 B
#define YP_OFF  1036288u      // bf16 [BATCH][DIM]     = 8,388,608 B
#define TL_OFF  9424896u      // f32  [BATCH]          =    32,768 B
#define PE_OFF  9457664u      // f32  [BATCH]          =    32,768 B

__device__ __forceinline__ unsigned short f2bf(float f) {
    unsigned int u = __float_as_uint(f);
    u += 0x7fffu + ((u >> 16) & 1u);           // round-to-nearest-even
    return (unsigned short)(u >> 16);
}

// log(expected_count(c)) for log-uniform sampler
__device__ __forceinline__ float logE(int c) {
    float cf = (float)c;
    float p = (logf(cf + 2.0f) - logf(cf + 1.0f)) * (1.0f / logf((float)NUM_CLASSES + 1.0f));
    float e = -expm1f((float)NUM_SAMPLED * log1pf(-p));
    return logf(e);
}

// TDM: DMA one 16-row x 512-col bf16 tile (rows contiguous, stride 512) into LDS,
// padding 16B per 1024B row (pad_interval=7 -> 1024B, pad_amount=3 -> 16B).
__device__ __forceinline__ void tdm_load_tile(const unsigned short* gsrc,
                                              unsigned lds_byte_addr) {
    unsigned long long ga = (unsigned long long)(uintptr_t)gsrc;
    v4u g0 = { 1u,                                   // count=1, no gather
               lds_byte_addr,                        // lds_addr
               (unsigned)(ga & 0xffffffffu),         // global_addr[31:0]
               (unsigned)((ga >> 32) & 0x1ffffffu) | 0x80000000u }; // [56:32] | type=2
    v8i g1 = { (int)0x07D10000,   // data_size=2B, pad_enable, pad_interval=7, pad_amount=3
               (int)0x02000000,   // tensor_dim0 = 512 (low16 << 16)
               (int)0x00100000,   // tensor_dim1 = 16  (low16 << 16)
               (int)0x02000000,   // tile_dim0 = 512   (<< 16)
               16,                // tile_dim1 = 16
               512,               // tensor_dim0_stride = 512
               0, 0 };
    v4i z4 = { 0, 0, 0, 0 };
    v8i z8 = { 0, 0, 0, 0, 0, 0, 0, 0 };
    __builtin_amdgcn_tensor_load_to_lds(g0, g1, z4, z4, z8, 0);
}

// --- K1: gather W[sampled] -> bf16 (padded), compute sb[n] = b[s] - logE(s) ---
__global__ __launch_bounds__(256) void k_prep_sampled(
    const int* __restrict__ sampled, const float* __restrict__ W,
    const float* __restrict__ b, unsigned short* __restrict__ wsb,
    float* __restrict__ sb) {
    int n = blockIdx.x;                 // 0..NPAD-1
    int t = threadIdx.x;
    if (n < NUM_SAMPLED) {
        int s = sampled[n];
        const float* src = W + (size_t)s * DIM;
        unsigned short* dst = wsb + (size_t)n * DIM;
        dst[t]       = f2bf(src[t]);
        dst[t + 256] = f2bf(src[t + 256]);
        if (t == 0) sb[n] = b[s] - logE(s);
    } else {
        unsigned short* dst = wsb + (size_t)n * DIM;
        dst[t] = 0; dst[t + 256] = 0;
        if (t == 0) sb[n] = -1e30f;     // padding column -> exp() == 0
    }
}

// --- K2: y_pred f32 -> bf16 ---
__global__ __launch_bounds__(256) void k_cvt_ypred(
    const float* __restrict__ yp32, unsigned short* __restrict__ yp16) {
    int idx = (blockIdx.x * 256 + threadIdx.x) * 4;
    float4 f = *(const float4*)(yp32 + idx);
    ushort4 o;
    o.x = f2bf(f.x); o.y = f2bf(f.y); o.z = f2bf(f.z); o.w = f2bf(f.w);
    *(ushort4*)(yp16 + idx) = o;
}

// --- K3: true logits, one wave per row, fp32 dot ---
__global__ __launch_bounds__(256) void k_true_logits(
    const int* __restrict__ y_true, const float* __restrict__ yp32,
    const float* __restrict__ W, const float* __restrict__ b,
    float* __restrict__ tl) {
    int wave = threadIdx.x >> 5;
    int lane = threadIdx.x & 31;
    int row = blockIdx.x * 8 + wave;
    int lab = y_true[row];
    const float* a = yp32 + (size_t)row * DIM;
    const float* w = W + (size_t)lab * DIM;
    float acc = 0.0f;
#pragma unroll
    for (int j = 0; j < DIM / 32; ++j) {
        int k = j * 32 + lane;
        acc = fmaf(a[k], w[k], acc);
    }
#pragma unroll
    for (int m = 16; m >= 1; m >>= 1) acc += __shfl_xor(acc, m, 32);
    if (lane == 0) tl[row] = acc + b[lab] - logE(lab);
}

// --- K4: WMMA GEMM fused with online logsumexp; B tiles TDM-staged in LDS ---
__global__ __launch_bounds__(256) void k_gemm_lse(
    const unsigned short* __restrict__ wsb, const float* __restrict__ sb,
    const unsigned short* __restrict__ yp16, const float* __restrict__ tl,
    const int* __restrict__ y_true, const int* __restrict__ sampled,
    float* __restrict__ pe) {
    __shared__ __align__(16) unsigned short lds_b[2][16 * LDS_ROW];

    int wave = threadIdx.x >> 5;
    int lane = threadIdx.x & 31;
    int half = lane >> 4;               // K-half selector (contiguous k halves, same for A and B)
    int lr   = lane & 15;
    int m0 = (blockIdx.x * 8 + wave) * 16;

    unsigned lds0 = (unsigned)(uintptr_t)(void*)&lds_b[0][0];
    unsigned lds1 = (unsigned)(uintptr_t)(void*)&lds_b[1][0];

    // Preload A tile: lane holds row m0+lr; per 32-K step, bf16 K slots
    // [k0+half*16, +16) (contiguous, matching B)
    v16bf A[16];
    const unsigned short* arow = yp16 + (size_t)(m0 + lr) * DIM;
#pragma unroll
    for (int kk = 0; kk < 16; ++kk) {
        int k0 = kk * 32;
        union { v16bf v; uint4 q[2]; } u;
        u.q[0] = *(const uint4*)(arow + k0 + half * 16);
        u.q[1] = *(const uint4*)(arow + k0 + half * 16 + 8);
        A[kk] = u.v;
    }

    int lab[8];
#pragma unroll
    for (int r = 0; r < 8; ++r) lab[r] = y_true[m0 + half * 8 + r];

    float rm[8], rs[8];
#pragma unroll
    for (int r = 0; r < 8; ++r) { rm[r] = -3.0e38f; rs[r] = 0.0f; }

    // prologue: DMA tile 0 into buffer 0 (wave 0 only)
    if (threadIdx.x < 32) tdm_load_tile(wsb, lds0);

    for (int nt = 0; nt < NTILES; ++nt) {
        if (threadIdx.x < 32) {
            if (nt + 1 < NTILES) {
                tdm_load_tile(wsb + (size_t)(nt + 1) * 16 * DIM,
                              ((nt + 1) & 1) ? lds1 : lds0);
                __builtin_amdgcn_s_wait_tensorcnt(1);   // tile nt landed
            } else {
                __builtin_amdgcn_s_wait_tensorcnt(0);   // last tile landed
            }
        }
        __syncthreads();

        const unsigned short* bcol = &lds_b[nt & 1][lr * LDS_ROW];
        v8f acc = {};
        // even/odd double-buffered B registers: loads stay one K-step ahead
        union BU { v16bf v; uint4 q[2]; };
        BU b0, b1;
        b0.q[0] = *(const uint4*)(bcol + half * 16);
        b0.q[1] = *(const uint4*)(bcol + half * 16 + 8);
#pragma unroll
        for (int kk = 0; kk < 16; kk += 2) {
            int k1 = (kk + 1) * 32;
            b1.q[0] = *(const uint4*)(bcol + k1 + half * 16);
            b1.q[1] = *(const uint4*)(bcol + k1 + half * 16 + 8);
            acc = __builtin_amdgcn_wmma_f32_16x16x32_bf16(
                false, A[kk], false, b0.v, (short)0, acc, false, false);
            if (kk + 2 < 16) {
                int k2 = (kk + 2) * 32;
                b0.q[0] = *(const uint4*)(bcol + k2 + half * 16);
                b0.q[1] = *(const uint4*)(bcol + k2 + half * 16 + 8);
            }
            acc = __builtin_amdgcn_wmma_f32_16x16x32_bf16(
                false, A[kk + 1], false, b1.v, (short)0, acc, false, false);
        }
#if __has_builtin(__builtin_amdgcn_sched_group_barrier)
        // pin pipeline: 4 DS-reads ahead, then {1 WMMA, 2 DS-reads} x15, last WMMA
        __builtin_amdgcn_sched_group_barrier(0x100, 4, 0);
#pragma unroll
        for (int i = 0; i < 15; ++i) {
            __builtin_amdgcn_sched_group_barrier(0x008, 1, 0);
            __builtin_amdgcn_sched_group_barrier(0x100, 2, 0);
        }
        __builtin_amdgcn_sched_group_barrier(0x008, 1, 0);
#endif

        int n = nt * 16 + lr;
        int   sn  = (n < NUM_SAMPLED) ? sampled[n] : -1;
        float sbv = sb[n];
#pragma unroll
        for (int r = 0; r < 8; ++r) {
            float x = acc[r] + sbv;
            if (sn == lab[r]) x -= 1000000000.0f;   // hit mask
            float mx = x;
#pragma unroll
            for (int m = 1; m <= 8; m <<= 1) mx = fmaxf(mx, __shfl_xor(mx, m, 32));
            float e = __expf(x - mx);
#pragma unroll
            for (int m = 1; m <= 8; m <<= 1) e += __shfl_xor(e, m, 32);
            float nm = fmaxf(rm[r], mx);
            rs[r] = rs[r] * __expf(rm[r] - nm) + e * __expf(mx - nm);
            rm[r] = nm;
        }
        __syncthreads();   // all waves done with this buffer before it is re-filled
    }

    // finalize: lanes 0-7 of each half own one row each
#pragma unroll
    for (int r = 0; r < 8; ++r) {
        if (lr == r) {
            int row = m0 + half * 8 + r;
            float t = tl[row];
            float nm = fmaxf(rm[r], t);
            float tot = rs[r] * __expf(rm[r] - nm) + __expf(t - nm);
            pe[row] = nm + logf(tot) - t;
        }
    }
}

// --- K5: deterministic mean reduction ---
__global__ __launch_bounds__(256) void k_reduce_mean(
    const float* __restrict__ pe, float* __restrict__ out) {
    __shared__ float sm[256];
    float s = 0.0f;
    for (int j = threadIdx.x; j < BATCH; j += 256) s += pe[j];
    sm[threadIdx.x] = s;
    __syncthreads();
    for (int st = 128; st > 0; st >>= 1) {
        if (threadIdx.x < st) sm[threadIdx.x] += sm[threadIdx.x + st];
        __syncthreads();
    }
    if (threadIdx.x == 0) out[0] = sm[0] * (1.0f / (float)BATCH);
}

extern "C" void kernel_launch(void* const* d_in, const int* in_sizes, int n_in,
                              void* d_out, int out_size, void* d_ws, size_t ws_size,
                              hipStream_t stream) {
    const int*   y_true  = (const int*)d_in[0];
    const float* y_pred  = (const float*)d_in[1];
    const float* W       = (const float*)d_in[2];
    const float* b       = (const float*)d_in[3];
    const int*   sampled = (const int*)d_in[4];
    float* out = (float*)d_out;

    char* ws = (char*)d_ws;
    unsigned short* wsb  = (unsigned short*)(ws + WSB_OFF);
    float*          sb   = (float*)(ws + SB_OFF);
    unsigned short* yp16 = (unsigned short*)(ws + YP_OFF);
    float*          tl   = (float*)(ws + TL_OFF);
    float*          pe   = (float*)(ws + PE_OFF);

    k_prep_sampled<<<NPAD, 256, 0, stream>>>(sampled, W, b, wsb, sb);
    k_cvt_ypred<<<(BATCH * DIM) / (256 * 4), 256, 0, stream>>>(y_pred, yp16);
    k_true_logits<<<BATCH / 8, 256, 0, stream>>>(y_true, y_pred, W, b, tl);
    k_gemm_lse<<<BATCH / 128, 256, 0, stream>>>(wsb, sb, yp16, tl, y_true, sampled, pe);
    k_reduce_mean<<<1, 256, 0, stream>>>(pe, out);
}
